// MyModel_87522843561002
// MI455X (gfx1250) — matrix operand order repaired and checked by reading, not compiled, not run on hardware
//
#include <hip/hip_runtime.h>
#include <math.h>

// ---------------------------------------------------------------------------
// Fused GRU scan for MI455X (gfx1250, wave32, WMMA bf16).
//
// Grid: 64 blocks, one per 16-row batch tile.  Block: 224 threads = 7 waves,
// wave jj owns hidden-unit tile [16*jj, 16*jj+16) of the 112-padded state.
// Weights live in LDS as pre-swizzled WMMA B fragments (bf16).
// ---------------------------------------------------------------------------

typedef __attribute__((ext_vector_type(16))) __bf16 v16bf;
typedef __attribute__((ext_vector_type(8)))  float  v8f;

#define UNITS    100
#define G3       (3*UNITS)
#define TSTEPS   512
#define FEAT     128
#define BROWS    16            // batch rows per block (one WMMA M tile)
#define NWAVE    7             // 7*16 = 112 >= UNITS
#define NTHREADS (NWAVE*32)

#define FRAG_ELEMS  512                          // 32(K) x 16(N) bf16
#define NFRAG       (2*3*7*4)                    // {Wk,Wr} x gate x jj x kc
#define WFRAG_ELEMS (NFRAG*FRAG_ELEMS)           // 86016 bf16
#define XROW_STRIDE 136                          // 128 + 8 pad (LDS banks)
#define XBUF_ELEMS  (BROWS*XROW_STRIDE)          // 2176 bf16
#define SMEM_ELEMS  (WFRAG_ELEMS + 2*XBUF_ELEMS) // 90368 bf16
#define SMEM_BYTES  (SMEM_ELEMS*2)               // 180736 B  (< 320 KB/WGP)

#define FRAGOFF(m,g,jj,kc) (((((m)*3+(g))*7+(jj))*4+(kc))*FRAG_ELEMS)

// float <-> bf16 without relying on __bf16 arithmetic conversions
__device__ __forceinline__ __bf16 f2bf(float f) {
  union { float f; unsigned u; } x; x.f = f;
  unsigned r = (x.u + 0x7fffu + ((x.u >> 16) & 1u)) >> 16;
  unsigned short s = (unsigned short)r;
  return *reinterpret_cast<__bf16*>(&s);
}
__device__ __forceinline__ float bf2f(__bf16 b) {
  unsigned short s = *reinterpret_cast<unsigned short*>(&b);
  union { unsigned u; float f; } x; x.u = ((unsigned)s) << 16;
  return x.f;
}

// B fragment: section0 = halves 0..7 of each lane (lane*16B), section1 =
// halves 8..15 (512B + lane*16B).  Two conflict-free ds_load_b128 per lane.
__device__ __forceinline__ v16bf load_frag(const __bf16* f, int lane) {
  v16bf v;
  reinterpret_cast<uint4*>(&v)[0] = reinterpret_cast<const uint4*>(f)[lane];
  reinterpret_cast<uint4*>(&v)[1] = reinterpret_cast<const uint4*>(f)[32 + lane];
  return v;
}

// A fragment per ISA 16-bit A layout: lane (hi,m) holds K = kc*32+hi*8+{0..7}
// in halves 0..7 and K = kc*32+hi*8+{16..23} in halves 8..15.
__device__ __forceinline__ v16bf load_a(const __bf16* row, int kc, int hi) {
  const __bf16* p = row + kc*32 + hi*8;
  v16bf v;
  reinterpret_cast<uint4*>(&v)[0] = *reinterpret_cast<const uint4*>(p);
  reinterpret_cast<uint4*>(&v)[1] = *reinterpret_cast<const uint4*>(p + 16);
  return v;
}

__device__ __forceinline__ v8f wmma_bf16(v16bf a, v16bf b, v8f c) {
  return __builtin_amdgcn_wmma_f32_16x16x32_bf16(
      /*neg_a=*/false, a, /*neg_b=*/false, b,
      /*c_mod=*/(short)0, c, /*reuse_a=*/false, /*reuse_b=*/false);
}

extern "C" __global__ __launch_bounds__(NTHREADS)
void gru_fused(const float* __restrict__ inputs,   // [1024,512,128]
               const float* __restrict__ kernel,   // [128,300]
               const float* __restrict__ rkernel,  // [100,300]
               const float* __restrict__ bias_in,  // [300]
               const float* __restrict__ bias_rec, // [300]
               const float* __restrict__ dense_w,  // [100]
               const float* __restrict__ dense_b,  // [1]
               float* __restrict__ out)            // [1024]
{
  extern __shared__ __bf16 smem[];
  __bf16* wfrag = smem;                 // 86016: weight B-fragments
  __bf16* xbuf  = smem + WFRAG_ELEMS;   //  2176: x tile (bf16)
  __bf16* hbuf  = xbuf + XBUF_ELEMS;    //  2176: h state (bf16)

  const int tid  = threadIdx.x;
  const int wave = tid >> 5;
  const int lane = tid & 31;
  const int hi   = lane >> 4;
  const int nn_l = lane & 15;
  const int jj   = wave;                // hidden-unit tile owned by this wave
  const int b0   = blockIdx.x * BROWS;

  // ---- pack kernel/rkernel into padded, per-gate WMMA B fragments ----
  for (int e = tid; e < WFRAG_ELEMS; e += NTHREADS) {
    int frag = e >> 9;
    int idx  = e & 511;
    int kk = idx >> 4, nn = idx & 15;
    int kc  = frag & 3;
    int t1  = frag >> 2;
    int fjj = t1 % 7; t1 /= 7;
    int g   = t1 % 3;
    int m   = t1 / 3;                   // 0 = kernel, 1 = rkernel
    int k = kc*32 + kk;
    int u = fjj*16 + nn;
    float val = 0.f;
    if (u < UNITS) {
      int c = g*UNITS + u;
      if (m == 0)          val = kernel[k*G3 + c];
      else if (k < UNITS)  val = rkernel[k*G3 + c];
    }
    int fl   = ((kk >> 3) & 1)*16 + nn; // lane holding (kk, nn)
    int half = (kk & 7) + ((kk >> 4) << 3); // half index within that lane
    int off  = (half < 8) ? (fl*8 + half) : (256 + fl*8 + (half - 8));
    wfrag[frag*FRAG_ELEMS + off] = f2bf(val);
  }
  for (int e = tid; e < XBUF_ELEMS; e += NTHREADS) hbuf[e] = f2bf(0.f);

  // ---- per-lane gate biases (broadcast over M) ----
  float bZ = 0.f, bR = 0.f, bXh = 0.f, bRh = 0.f;
  { int u = jj*16 + nn_l;
    if (u < UNITS) {
      bZ  = bias_in[u]          + bias_rec[u];
      bR  = bias_in[UNITS + u]  + bias_rec[UNITS + u];
      bXh = bias_in[2*UNITS + u];          // candidate keeps biases separate:
      bRh = bias_rec[2*UNITS + u];         // hh = tanh(xh + r * ih)
    } }

  float hold[8];
  #pragma unroll
  for (int v = 0; v < 8; ++v) hold[v] = 0.f;

  __syncthreads();

  for (int t = 0; t < TSTEPS; ++t) {
    // stage x tile: 16 rows x 128 f32 -> bf16 LDS (coalesced)
    for (int e = tid; e < BROWS*FEAT; e += NTHREADS) {
      int r = e >> 7, f = e & 127;
      xbuf[r*XROW_STRIDE + f] =
          f2bf(inputs[((size_t)(b0 + r)*TSTEPS + t)*FEAT + f]);
    }
    // prefetch next step's rows (global_prefetch_b8, 128B lines)
    if (t + 1 < TSTEPS && tid < 64)
      __builtin_prefetch(
          &inputs[((size_t)(b0 + (tid >> 2))*TSTEPS + (t + 1))*FEAT + (tid & 3)*32],
          0, 1);
    __syncthreads();                    // x ready; prev-step h writes visible

    v16bf xA[4], hA[4];
    const __bf16* xrow = xbuf + nn_l*XROW_STRIDE;
    const __bf16* hrow = hbuf + nn_l*XROW_STRIDE;
    #pragma unroll
    for (int kc = 0; kc < 4; ++kc) {
      xA[kc] = load_a(xrow, kc, hi);
      hA[kc] = load_a(hrow, kc, hi);
    }

    v8f accZ, accR, accXh, accRh;
    #pragma unroll
    for (int v = 0; v < 8; ++v) { accZ[v]=bZ; accR[v]=bR; accXh[v]=bXh; accRh[v]=bRh; }

    #pragma unroll
    for (int kc = 0; kc < 4; ++kc)
      accZ  = wmma_bf16(xA[kc], load_frag(wfrag + FRAGOFF(0,0,jj,kc), lane), accZ);
    #pragma unroll
    for (int kc = 0; kc < 4; ++kc)
      accZ  = wmma_bf16(hA[kc], load_frag(wfrag + FRAGOFF(1,0,jj,kc), lane), accZ);
    #pragma unroll
    for (int kc = 0; kc < 4; ++kc)
      accR  = wmma_bf16(xA[kc], load_frag(wfrag + FRAGOFF(0,1,jj,kc), lane), accR);
    #pragma unroll
    for (int kc = 0; kc < 4; ++kc)
      accR  = wmma_bf16(hA[kc], load_frag(wfrag + FRAGOFF(1,1,jj,kc), lane), accR);
    #pragma unroll
    for (int kc = 0; kc < 4; ++kc)
      accXh = wmma_bf16(xA[kc], load_frag(wfrag + FRAGOFF(0,2,jj,kc), lane), accXh);
    #pragma unroll
    for (int kc = 0; kc < 4; ++kc)
      accRh = wmma_bf16(hA[kc], load_frag(wfrag + FRAGOFF(1,2,jj,kc), lane), accRh);

    __syncthreads();                    // everyone finished reading hbuf

    #pragma unroll
    for (int v = 0; v < 8; ++v) {       // C layout: M = v + hi*8, N = nn_l
      float z  = 1.f / (1.f + __expf(-accZ[v]));
      float rr = 1.f / (1.f + __expf(-accR[v]));
      float hh = tanhf(accXh[v] + rr*accRh[v]);
      float hn = z*hold[v] + (1.f - z)*hh;
      hold[v]  = hn;
      hbuf[(v + hi*8)*XROW_STRIDE + jj*16 + nn_l] = f2bf(hn);
    }
  }
  __syncthreads();

  // Dense(1): out[b] = h . w + b
  if (tid < BROWS) {
    float s = dense_b[0];
    for (int u = 0; u < UNITS; ++u)
      s += bf2f(hbuf[tid*XROW_STRIDE + u]) * dense_w[u];
    out[b0 + tid] = s;
  }
}

extern "C" void kernel_launch(void* const* d_in, const int* in_sizes, int n_in,
                              void* d_out, int out_size, void* d_ws, size_t ws_size,
                              hipStream_t stream) {
  (void)in_sizes; (void)n_in; (void)out_size; (void)d_ws; (void)ws_size;
  hipFuncSetAttribute(reinterpret_cast<const void*>(gru_fused),
                      hipFuncAttributeMaxDynamicSharedMemorySize, SMEM_BYTES);
  gru_fused<<<dim3(1024 / BROWS), dim3(NTHREADS), SMEM_BYTES, stream>>>(
      (const float*)d_in[0], (const float*)d_in[1], (const float*)d_in[2],
      (const float*)d_in[3], (const float*)d_in[4], (const float*)d_in[5],
      (const float*)d_in[6], (float*)d_out);
}
